// TopologicalLoss_23175643529655
// MI455X (gfx1250) — matrix operand order repaired and checked by reading, not compiled, not run on hardware
//
#include <hip/hip_runtime.h>

// ---------------------------------------------------------------------------
// Topological loss for MI455X (gfx1250, wave32).
//   predictions: (4,4,512,512) f32   targets: (4,512,512) i32   out: (4,) f32
// Pipeline: argmax masks -> morphology pyramid (WMMA 3x3 box filter, tiles
//           staged with global_load_async_to_lds_b32) -> label-equivalence
//           CCL -> component sizes -> top-64 -> L1 loss.
// ---------------------------------------------------------------------------

#define IMG_H 512
#define IMG_W 512
#define HWN   (IMG_H * IMG_W)   // 262144
#define NB    4
#define NC    4
#define NIMG  (NB * NC)         // 16
#define SENT  HWN
#define CAND_MAX 131072         // max 4-connected components in 512x512

typedef __attribute__((ext_vector_type(16))) _Float16 v16h;
typedef __attribute__((ext_vector_type(8)))  float    v8f;

static __device__ __forceinline__ int imin(int a, int b) { return a < b ? a : b; }
static __device__ __forceinline__ int iabs(int a) { return a < 0 ? -a : a; }

// ---------------------------------------------------------------------------
// 1) argmax over classes -> one-hot f16 masks for predictions and targets
// ---------------------------------------------------------------------------
__global__ __launch_bounds__(256)
void build_masks_kernel(const float* __restrict__ preds,
                        const int* __restrict__ targets,
                        _Float16* __restrict__ maskP,
                        _Float16* __restrict__ maskT)
{
    int t = blockIdx.x * blockDim.x + threadIdx.x;
    if (t >= NB * HWN) return;
    int b = t / HWN;
    int p = t - b * HWN;
    const float* base = preds + (size_t)b * NC * HWN + p;
    __builtin_prefetch(base, 0, 0);            // global_prefetch_b8
    float best = base[0];
    int   bi   = 0;
    #pragma unroll
    for (int c = 1; c < NC; ++c) {
        float v = base[(size_t)c * HWN];
        if (v > best) { best = v; bi = c; }    // first-max wins (matches argmax)
    }
    int tv = targets[t];
    #pragma unroll
    for (int c = 0; c < NC; ++c) {
        size_t o = ((size_t)b * NC + c) * HWN + p;
        maskP[o] = (_Float16)(c == bi ? 1.0f : 0.0f);
        maskT[o] = (_Float16)(c == tv ? 1.0f : 0.0f);
    }
}

// ---------------------------------------------------------------------------
// 2) 3x3 box filter via two chained V_WMMA_F32_16X16X32_F16, then threshold.
//    box = Mv * X * Mh with halos folded into the K=32 dimension.
//    Tile interior staged Global->LDS with async ops (ASYNCcnt).
//    One wave per 16x16 output tile. mode 0 = erode (==9), 1 = dilate (>0).
// ---------------------------------------------------------------------------
__global__ __launch_bounds__(256)
void box_morph_kernel(const _Float16* __restrict__ src,
                      _Float16* __restrict__ dst, int mode)
{
    // per-wave: [0..255] async-staged 16x16 source tile, [256..511] P relayout
    __shared__ _Float16 lds[8][512];
    int wave = threadIdx.x >> 5;
    int lane = threadIdx.x & 31;
    int tile = blockIdx.x * 8 + wave;          // 2048 blocks * 8 waves = 16384 tiles
    int img  = tile >> 10;                     // 1024 tiles per image
    int trow = (tile >> 5) & 31;
    int tcol = tile & 31;
    const _Float16* im = src + (size_t)img * HWN;
    int y0 = trow * 16, x0 = tcol * 16;
    int hh = lane >> 4;                        // lane half
    int ln = lane & 15;

    // ---- async-stage the 16x16 tile interior into LDS:
    //      4 x global_load_async_to_lds_b32, each moving 4 rows (32 lanes x 4B)
    _Float16* stage = &lds[wave][0];           // row-major 16x16 halfs (512 B)
    {
        int sub   = lane & 7;                  // dword within row (0..7)
        int rbase = lane >> 3;                 // 0..3
        const char* gtile = (const char*)(im + (size_t)y0 * IMG_W + x0);
        #pragma unroll
        for (int j = 0; j < 4; ++j) {
            int row = j * 4 + rbase;
            unsigned long long ga =
                (unsigned long long)(uintptr_t)(gtile + row * (IMG_W * 2) + sub * 4);
            unsigned la =
                (unsigned)(uintptr_t)((char*)stage + row * 32 + sub * 4);
            asm volatile("global_load_async_to_lds_b32 %0, %1, off"
                         :: "v"(la), "v"(ga) : "memory");
        }
        asm volatile("s_wait_asynccnt 0x0" ::: "memory");
    }

    // ---- B operand: X_ext (32x16). Rows 0..15 = tile (from LDS stage),
    //      row 16 = top halo, 17 = bottom halo, 18..31 = 0.
    //      b[idx] = Xext[k(idx)][ln], k(idx) = idx<8 ? hh*8+idx : 16+hh*8+(idx-8)
    v16h bmat;
    #pragma unroll
    for (int idx = 0; idx < 16; ++idx) {
        int k = (idx < 8) ? (hh * 8 + idx) : (16 + hh * 8 + (idx - 8));
        float v = 0.0f;
        if (k < 16) {
            v = (float)stage[k * 16 + ln];
        } else if (k == 16 || k == 17) {
            int y = (k == 16) ? (y0 - 1) : (y0 + 16);
            if (y >= 0 && y < IMG_H) v = (float)im[y * IMG_W + x0 + ln];
        }
        bmat[idx] = (_Float16)v;
    }

    // ---- A operand: Mv_ext (16x32). a[idx] = A[m=ln][k(idx)],
    //      k(idx) = idx<8 ? hh*8+idx : 16+hh*8+(idx-8)
    v16h amat;
    #pragma unroll
    for (int idx = 0; idx < 16; ++idx) {
        int k = (idx < 8) ? (hh * 8 + idx) : (16 + hh * 8 + (idx - 8));
        float v = 0.0f;
        if (k < 16)       v = (iabs(k - ln) <= 1) ? 1.0f : 0.0f;
        else if (k == 16) v = (ln == 0)  ? 1.0f : 0.0f;   // top halo row select
        else if (k == 17) v = (ln == 15) ? 1.0f : 0.0f;   // bottom halo row select
        amat[idx] = (_Float16)v;
    }

    v8f zero = {};
    // Vertical pass (with halos): P[i][j] = X[y0+i-1]+X[y0+i]+X[y0+i+1] at col j
    v8f P = __builtin_amdgcn_wmma_f32_16x16x32_f16(false, amat, false, bmat,
                                                   (short)0, zero, false, false);

    // ---- Re-distribute P (C/D layout) -> A layout via LDS (f16, exact: sums<=3)
    _Float16* tl = &lds[wave][256];
    #pragma unroll
    for (int v = 0; v < 8; ++v)
        tl[(hh * 8 + v) * 16 + ln] = (_Float16)P[v];
    __syncthreads();

    v16h a2;
    #pragma unroll
    for (int idx = 0; idx < 16; ++idx) {
        int k = (idx < 8) ? (hh * 8 + idx) : (16 + hh * 8 + (idx - 8));
        float v = 0.0f;
        if (k < 16) {
            v = (float)tl[ln * 16 + k];                 // P[m][k]
        } else if (k == 16 || k == 17) {                // halo columns, vert-summed
            int xh = (k == 16) ? (x0 - 1) : (x0 + 16);
            if (xh >= 0 && xh < IMG_W) {
                int yc = y0 + ln;
                #pragma unroll
                for (int dy = -1; dy <= 1; ++dy) {
                    int yy = yc + dy;
                    if (yy >= 0 && yy < IMG_H) v += (float)im[yy * IMG_W + xh];
                }
            }
        }
        a2[idx] = (_Float16)v;
    }

    // ---- B operand: Mh_ext (32x16). b2[idx] = Mh_ext[k(idx)][n=ln], table k order
    v16h b2;
    #pragma unroll
    for (int idx = 0; idx < 16; ++idx) {
        int k = (idx < 8) ? (hh * 8 + idx) : (16 + hh * 8 + (idx - 8));
        float v = 0.0f;
        if (k < 16)       v = (iabs(k - ln) <= 1) ? 1.0f : 0.0f;
        else if (k == 16) v = (ln == 0)  ? 1.0f : 0.0f;   // left halo column
        else if (k == 17) v = (ln == 15) ? 1.0f : 0.0f;   // right halo column
        b2[idx] = (_Float16)v;
    }

    // Horizontal pass: Q = full 3x3 box sum (exact integers 0..9)
    v8f Q = __builtin_amdgcn_wmma_f32_16x16x32_f16(false, a2, false, b2,
                                                   (short)0, zero, false, false);

    _Float16* out = dst + (size_t)img * HWN;
    #pragma unroll
    for (int v = 0; v < 8; ++v) {
        float q = Q[v];
        float r = (mode == 0) ? (q > 8.5f ? 1.0f : 0.0f)   // erode: sum == 9
                              : (q > 0.5f ? 1.0f : 0.0f);  // dilate: sum > 0
        out[(y0 + hh * 8 + v) * IMG_W + (x0 + ln)] = (_Float16)r;
    }
}

// ---------------------------------------------------------------------------
// 3) Connected components: label-equivalence with root chasing + atomicMin.
// ---------------------------------------------------------------------------
__global__ __launch_bounds__(256)
void ccl_init_kernel(const _Float16* __restrict__ img, int* __restrict__ labels)
{
    int t = blockIdx.x * blockDim.x + threadIdx.x;
    if (t >= NIMG * HWN) return;
    int p = t & (HWN - 1);
    bool fg = (float)img[t] > 0.5f;
    labels[t] = fg ? p : SENT;
}

__global__ __launch_bounds__(256)
void ccl_merge_kernel(const _Float16* __restrict__ img, int* __restrict__ labels)
{
    int t = blockIdx.x * blockDim.x + threadIdx.x;
    if (t >= NIMG * HWN) return;
    if ((float)img[t] <= 0.5f) return;
    int im = t >> 18;                          // / HWN
    int p  = t & (HWN - 1);
    int* L = labels + (size_t)im * HWN;
    const _Float16* X = img + (size_t)im * HWN;
    int y = p >> 9, x = p & (IMG_W - 1);

    int m = L[p];
    if (y > 0         && (float)X[p - IMG_W] > 0.5f) m = imin(m, L[p - IMG_W]);
    if (y < IMG_H - 1 && (float)X[p + IMG_W] > 0.5f) m = imin(m, L[p + IMG_W]);
    if (x > 0         && (float)X[p - 1]     > 0.5f) m = imin(m, L[p - 1]);
    if (x < IMG_W - 1 && (float)X[p + 1]     > 0.5f) m = imin(m, L[p + 1]);

    // chase m and own root (labels only decrease; strictly descending -> bounded)
    for (;;) { int u = L[m]; if (u >= m) break; m = u; }
    int r = p;
    for (;;) { int u = L[r]; if (u >= r) break; r = u; }
    if (m < r) atomicMin(&L[r], m);
}

__global__ __launch_bounds__(256)
void ccl_resolve_kernel(const _Float16* __restrict__ img, int* __restrict__ labels)
{
    int t = blockIdx.x * blockDim.x + threadIdx.x;
    if (t >= NIMG * HWN) return;
    if ((float)img[t] <= 0.5f) return;
    int im = t >> 18;
    int p  = t & (HWN - 1);
    int* L = labels + (size_t)im * HWN;
    int r = p;
    for (;;) { int u = L[r]; if (u >= r) break; r = u; }
    L[p] = r;
}

// ---------------------------------------------------------------------------
// 4) Component sizes + top-64 per image
// ---------------------------------------------------------------------------
__global__ __launch_bounds__(256)
void zero_counts_kernel(int* __restrict__ counts, int* __restrict__ ncand)
{
    int t = blockIdx.x * blockDim.x + threadIdx.x;
    if (t < NIMG * HWN) counts[t] = 0;
    if (t < NIMG) ncand[t] = 0;
}

__global__ __launch_bounds__(256)
void ccl_count_kernel(const _Float16* __restrict__ img,
                      const int* __restrict__ labels, int* __restrict__ counts)
{
    int t = blockIdx.x * blockDim.x + threadIdx.x;
    if (t >= NIMG * HWN) return;
    if ((float)img[t] <= 0.5f) return;
    int im = t >> 18;
    atomicAdd(&counts[(size_t)im * HWN + labels[t]], 1);
}

__global__ __launch_bounds__(256)
void compact_kernel(const int* __restrict__ counts, int* __restrict__ cand,
                    int* __restrict__ ncand)
{
    int t = blockIdx.x * blockDim.x + threadIdx.x;
    if (t >= NIMG * HWN) return;
    int c = counts[t];
    if (c > 0) {
        int im  = t >> 18;
        int pos = atomicAdd(&ncand[im], 1);
        if (pos < CAND_MAX) cand[(size_t)im * CAND_MAX + pos] = c;
    }
}

// one block per image: 64 deterministic max-extract passes (value<<32 | index)
__global__ __launch_bounds__(256)
void topk_select_kernel(int* __restrict__ cand, const int* __restrict__ ncand,
                        float* __restrict__ outBase)
{
    __shared__ unsigned long long red[256];
    int im  = blockIdx.x;
    int tid = threadIdx.x;
    int n   = imin(ncand[im], CAND_MAX);
    int* cd = cand + (size_t)im * CAND_MAX;
    float* out = outBase + im * 64;

    for (int kk = 0; kk < 64; ++kk) {
        unsigned long long best = 0ULL;
        for (int i = tid; i < n; i += 256) {
            int c = cd[i];
            if (c > 0) {
                unsigned long long key =
                    ((unsigned long long)(unsigned)c << 32) | (unsigned)i;
                if (key > best) best = key;
            }
        }
        red[tid] = best;
        __syncthreads();
        for (int s = 128; s > 0; s >>= 1) {
            if (tid < s && red[tid + s] > red[tid]) red[tid] = red[tid + s];
            __syncthreads();
        }
        if (tid == 0) {
            unsigned long long b = red[0];
            out[kk] = (float)(int)(b >> 32);
            if (b) cd[(int)(b & 0xffffffffULL)] = 0;   // remove extracted max
        }
        __syncthreads();
    }
}

// ---------------------------------------------------------------------------
// 5) loss[c] = sum_{lvl,b,k} |bar_p - bar_t|
// ---------------------------------------------------------------------------
__global__ void loss_kernel(const float* __restrict__ bar, float* __restrict__ out)
{
    int c = threadIdx.x;
    if (c >= NC) return;
    float acc = 0.0f;
    for (int lvl = 0; lvl < 5; ++lvl)
        for (int b = 0; b < NB; ++b) {
            int img = b * NC + c;
            const float* bp = bar + (((0 * 5 + lvl) * NIMG + img) * 64);
            const float* bt = bar + (((1 * 5 + lvl) * NIMG + img) * 64);
            for (int k = 0; k < 64; ++k) acc += fabsf(bp[k] - bt[k]);
        }
    out[c] = acc;
}

// ---------------------------------------------------------------------------
// Host orchestration
// ---------------------------------------------------------------------------
extern "C" void kernel_launch(void* const* d_in, const int* in_sizes, int n_in,
                              void* d_out, int out_size, void* d_ws, size_t ws_size,
                              hipStream_t stream)
{
    const float* preds   = (const float*)d_in[0];
    const int*   targets = (const int*)d_in[1];
    float*       out     = (float*)d_out;

    char* ws = (char*)d_ws;
    size_t off = 0;
    auto take = [&](size_t bytes) -> char* {
        char* p = ws + off;
        off = (off + bytes + 255) & ~(size_t)255;
        return p;
    };
    _Float16* maskP  = (_Float16*)take((size_t)NIMG * HWN * sizeof(_Float16)); // 8MB
    _Float16* maskT  = (_Float16*)take((size_t)NIMG * HWN * sizeof(_Float16)); // 8MB
    _Float16* bufA   = (_Float16*)take((size_t)NIMG * HWN * sizeof(_Float16)); // 8MB
    _Float16* bufB   = (_Float16*)take((size_t)NIMG * HWN * sizeof(_Float16)); // 8MB
    int*      labels = (int*)take((size_t)NIMG * HWN * sizeof(int));           // 16MB
    int*      counts = (int*)take((size_t)NIMG * HWN * sizeof(int));           // 16MB
    int*      cand   = (int*)take((size_t)NIMG * CAND_MAX * sizeof(int));      // 8MB
    int*      ncand  = (int*)take(NIMG * sizeof(int));
    float*    bar    = (float*)take((size_t)2 * 5 * NIMG * 64 * sizeof(float));

    const int GB = (NIMG * HWN) / 256;   // 16384 blocks for per-pixel kernels

    build_masks_kernel<<<(NB * HWN + 255) / 256, 256, 0, stream>>>(
        preds, targets, maskP, maskT);

    auto morph = [&](const _Float16* s, _Float16* d, int mode) {
        box_morph_kernel<<<2048, 256, 0, stream>>>(s, d, mode);
    };
    auto process = [&](const _Float16* img, int set, int lvl) {
        ccl_init_kernel<<<GB, 256, 0, stream>>>(img, labels);
        for (int it = 0; it < 12; ++it)
            ccl_merge_kernel<<<GB, 256, 0, stream>>>(img, labels);
        ccl_resolve_kernel<<<GB, 256, 0, stream>>>(img, labels);
        zero_counts_kernel<<<GB, 256, 0, stream>>>(counts, ncand);
        ccl_count_kernel<<<GB, 256, 0, stream>>>(img, labels, counts);
        compact_kernel<<<GB, 256, 0, stream>>>(counts, cand, ncand);
        topk_select_kernel<<<NIMG, 256, 0, stream>>>(
            cand, ncand, bar + ((size_t)(set * 5 + lvl) * NIMG) * 64);
    };

    for (int set = 0; set < 2; ++set) {
        const _Float16* base = (set == 0) ? maskP : maskT;
        process(base, set, 2);                       // original (mid)
        morph(base, bufA, 0); process(bufA, set, 3); // erode^1
        morph(bufA, bufB, 0); process(bufB, set, 4); // erode^2
        morph(base, bufA, 1); process(bufA, set, 1); // dilate^1
        morph(bufA, bufB, 1); process(bufB, set, 0); // dilate^2
    }

    loss_kernel<<<1, 32, 0, stream>>>(bar, out);
}
